// MHA_76149770158401
// MI455X (gfx1250) — matrix operand order repaired and checked by reading, not compiled, not run on hardware
//
#include <hip/hip_runtime.h>
#include <hip/hip_bf16.h>

// ---------------------------------------------------------------------------
// MHA for MI455X (gfx1250): bf16 WMMA everywhere, f32 accumulation.
// B=2, T=2048, D=1280, H=20, dh=64.  Compute-bound (~97 GFLOP vs ~70MB HBM),
// so all matmuls use v_wmma_f32_16x16x32_bf16.
// GEMM wave tile = 32x64 (2x4 WMMA): 1.5 b128 loads per WMMA, ~150 live VGPRs
// (the 64x64 variant spilled accumulators to scratch -> reverted).
// ---------------------------------------------------------------------------

typedef __bf16 bf16_t;
typedef __attribute__((ext_vector_type(16))) __bf16       v16bf;
typedef __attribute__((ext_vector_type(8)))  float        v8f;
typedef __attribute__((ext_vector_type(4)))  unsigned int v4u;

#define T_SEQ 2048
#define D_IN  1280
#define HEADS 20
#define DH    64
#define BATCH 2
#define MROWS (BATCH * T_SEQ)   // 4096

union FragU { v16bf f; v4u u[2]; };

// A-matrix fragment (16x32 bf16, one row per lane-of-16):
// lanes 0-15 hold K 0-15, lanes 16-31 hold K 16-31; 16 contiguous K per lane.
// p points at &row[kstep + (lane>>4)*16].
__device__ __forceinline__ v16bf load_frag_a(const bf16_t* p) {
    FragU r;
    r.u[0] = *(const v4u*)(p);
    r.u[1] = *(const v4u*)(p + 8);
    return r.f;
}

// B-matrix fragment (32x16 bf16) loaded from row-major [n][k] storage:
// lanes 0-15 hold K 0-15, lanes 16-31 hold K 16-31; 16 contiguous K per lane.
// p points at &Bt[n][kstep + (lane>>4)*16].
__device__ __forceinline__ v16bf load_frag_b(const bf16_t* p) {
    FragU r;
    r.u[0] = *(const v4u*)(p);
    r.u[1] = *(const v4u*)(p + 8);
    return r.f;
}

__device__ __forceinline__ v8f wmma_bf16(v16bf a, v16bf b, v8f c) {
    return __builtin_amdgcn_wmma_f32_16x16x32_bf16(
        /*neg_a=*/false, a, /*neg_b=*/false, b,
        /*c_mod=*/(short)0, c, /*reuse_a=*/false, /*reuse_b=*/false);
}

// ------------------------- conversion kernels ------------------------------

__global__ void cvt_f32_to_bf16(const float* __restrict__ in,
                                bf16_t* __restrict__ out, int n) {
    int i = (blockIdx.x * blockDim.x + threadIdx.x) * 4;
    if (i + 3 < n) {
        float4 v = *(const float4*)(in + i);
        out[i + 0] = (bf16_t)v.x;
        out[i + 1] = (bf16_t)v.y;
        out[i + 2] = (bf16_t)v.z;
        out[i + 3] = (bf16_t)v.w;
    }
}

// Wt[n*K + k] = (bf16) W[k*N + n]   (weights are [K][N] row-major in memory)
__global__ void transpose_to_bf16(const float* __restrict__ in,
                                  bf16_t* __restrict__ out, int N, int K) {
    int i = blockIdx.x * blockDim.x + threadIdx.x;
    if (i < N * K) {
        int n = i / K, k = i - n * K;
        out[i] = (bf16_t)in[(size_t)k * N + n];
    }
}

// ------------------------------- GEMM --------------------------------------
// C[M,N] = A[M,K](bf16, row-major) x Bt[N,K](bf16, row-major == B^T).
// Block: 256 threads = 8 waves laid out 4(M) x 2(N); wave tile 32x64 (2x4 WMMA).
// Block tile 128x128.  Grid: (N/128, M/128).
// mode 0: store bf16 into [B,H,T,dh]  (Q,K projections)
// mode 1: store bf16 into [B,H,dh,T]  (V projection, pre-transposed for PV)
// mode 2: store f32  row-major [M,N]  (final output projection)
__global__ void gemm_bf16_wmma(const bf16_t* __restrict__ A,
                               const bf16_t* __restrict__ Bt,
                               void* __restrict__ out,
                               int M, int N, int K, int mode) {
    const int lane = threadIdx.x & 31;
    const int w    = threadIdx.x >> 5;
    const int wm   = w & 3, wn = w >> 2;
    const int lr   = lane & 15, hi = lane >> 4;
    const int m0   = blockIdx.y * 128 + wm * 32;
    const int n0   = blockIdx.x * 128 + wn * 64;

    v8f acc[2][4] = {};

    const bf16_t* Ar[2];
    const bf16_t* Br[4];
#pragma unroll
    for (int i = 0; i < 2; ++i)
        Ar[i] = A + (size_t)(m0 + i * 16 + lr) * K + hi * 16;
#pragma unroll
    for (int j = 0; j < 4; ++j)
        Br[j] = Bt + (size_t)(n0 + j * 16 + lr) * K + hi * 16;

    for (int ks = 0; ks < K; ks += 32) {
        v16bf a[2], b[4];
#pragma unroll
        for (int i = 0; i < 2; ++i) a[i] = load_frag_a(Ar[i] + ks);
#pragma unroll
        for (int j = 0; j < 4; ++j) b[j] = load_frag_b(Br[j] + ks);
#pragma unroll
        for (int i = 0; i < 2; ++i)
#pragma unroll
            for (int j = 0; j < 4; ++j)
                acc[i][j] = wmma_bf16(a[i], b[j], acc[i][j]);
    }

    // C layout: lane holds col n = lane&15; VGPR r holds row r + 8*(lane>>4).
#pragma unroll
    for (int i = 0; i < 2; ++i) {
#pragma unroll
        for (int j = 0; j < 4; ++j) {
            int n     = n0 + j * 16 + lr;
            int mBase = m0 + i * 16 + hi * 8;
#pragma unroll
            for (int r = 0; r < 8; ++r) {
                float val = acc[i][j][r];
                int m = mBase + r;
                if (mode == 2) {
                    ((float*)out)[(size_t)m * N + n] = val;
                } else {
                    int b_ = m >> 11, t = m & (T_SEQ - 1);
                    int h  = n >> 6,  d = n & (DH - 1);
                    size_t idx;
                    if (mode == 0)       // [B,H,T,dh]
                        idx = ((((size_t)(b_ * HEADS + h)) << 11) + t) * DH + d;
                    else                 // [B,H,dh,T]
                        idx = ((((size_t)(b_ * HEADS + h)) * DH + d) << 11) + t;
                    ((bf16_t*)out)[idx] = (bf16_t)val;
                }
            }
        }
    }
}

// ---------------------------- attention ------------------------------------
// One wave owns 16 query rows for the full key loop (flash-style, f32 stats,
// log2-domain softmax so every exponential is a single v_exp_f32).
// Block = 256 threads = 8 waves => 128 queries/block.
// Grid = B*H * (T/128) = 640 blocks.
__global__ void attn_kernel(const bf16_t* __restrict__ Q,
                            const bf16_t* __restrict__ Kb,
                            const bf16_t* __restrict__ Vt,
                            bf16_t* __restrict__ Ctx) {
    __shared__ bf16_t plds[8 * 16 * 32];   // per-wave 16x32 bf16 P staging

    const int lane = threadIdx.x & 31;
    const int w    = threadIdx.x >> 5;
    const int lr   = lane & 15, hi = lane >> 4;
    const int bh   = blockIdx.x >> 4;      // b*HEADS + h
    const int q0   = (blockIdx.x & 15) * 128 + w * 16;

    const bf16_t* Qh = Q  + (size_t)bh * T_SEQ * DH;
    const bf16_t* Kh = Kb + (size_t)bh * T_SEQ * DH;
    const bf16_t* Vh = Vt + (size_t)bh * DH * T_SEQ;
    bf16_t* lp = plds + w * (16 * 32);

    // Q fragments are invariant across the key loop (A-matrix, 16x32 each).
    v16bf qa0 = load_frag_a(Qh + (size_t)(q0 + lr) * DH + 0  + hi * 16);
    v16bf qa1 = load_frag_a(Qh + (size_t)(q0 + lr) * DH + 32 + hi * 16);

    float mrow[8], lrow[8];
    v8f o0 = {}, o1 = {}, o2 = {}, o3 = {};
#pragma unroll
    for (int r = 0; r < 8; ++r) { mrow[r] = -1e30f; lrow[r] = 0.0f; }

    // 1/sqrt(dh) * log2(e): softmax runs in the log2 domain (exp2 only).
    const float scale = 0.125f * 1.44269504088896f;

    for (int kt = 0; kt < T_SEQ; kt += 32) {
        // S = Q @ K^T for 32 keys: two 16x16 tiles, dh=64 in two K-steps.
        v8f s0 = {}, s1 = {};
        {
            v16bf b;
            b  = load_frag_b(Kh + (size_t)(kt + lr)      * DH + 0  + hi * 16);
            s0 = wmma_bf16(qa0, b, s0);
            b  = load_frag_b(Kh + (size_t)(kt + lr)      * DH + 32 + hi * 16);
            s0 = wmma_bf16(qa1, b, s0);
            b  = load_frag_b(Kh + (size_t)(kt + 16 + lr) * DH + 0  + hi * 16);
            s1 = wmma_bf16(qa0, b, s1);
            b  = load_frag_b(Kh + (size_t)(kt + 16 + lr) * DH + 32 + hi * 16);
            s1 = wmma_bf16(qa1, b, s1);
        }

        // Online softmax. Row m lives in VGPR r across a 16-lane half, so
        // row reductions are shfl_xor with masks 1,2,4,8 (stays in-half).
        float p0[8], p1[8], alpha[8];
#pragma unroll
        for (int r = 0; r < 8; ++r) {
            float a  = s0[r] * scale;
            float bb = s1[r] * scale;
            float mx = fmaxf(a, bb);
#pragma unroll
            for (int msk = 1; msk < 16; msk <<= 1)
                mx = fmaxf(mx, __shfl_xor(mx, msk, 32));
            float mn = fmaxf(mrow[r], mx);
            alpha[r] = exp2f(mrow[r] - mn);
            mrow[r]  = mn;
            p0[r] = exp2f(a - mn);
            p1[r] = exp2f(bb - mn);
            float rs = p0[r] + p1[r];
#pragma unroll
            for (int msk = 1; msk < 16; msk <<= 1)
                rs += __shfl_xor(rs, msk, 32);
            lrow[r] = lrow[r] * alpha[r] + rs;
        }
#pragma unroll
        for (int r = 0; r < 8; ++r) {
            o0[r] *= alpha[r]; o1[r] *= alpha[r];
            o2[r] *= alpha[r]; o3[r] *= alpha[r];
        }

        // Transpose P from C-layout to A-layout through wave-private LDS.
#pragma unroll
        for (int r = 0; r < 8; ++r) {
            int row = r + hi * 8;
            lp[row * 32 + lr]      = (bf16_t)p0[r];
            lp[row * 32 + 16 + lr] = (bf16_t)p1[r];
        }
        asm volatile("s_wait_dscnt 0x0" ::: "memory");   // CDNA5 split counter
        v16bf pa = load_frag_a(lp + lr * 32 + hi * 16);
        asm volatile("s_wait_dscnt 0x0" ::: "memory");   // reads done before next-iter WAR

        // O += P @ V  (V pre-transposed: Vt[d][t], row-major in key dim).
        v16bf b;
        b  = load_frag_b(Vh + (size_t)(0  + lr) * T_SEQ + kt + hi * 16);
        o0 = wmma_bf16(pa, b, o0);
        b  = load_frag_b(Vh + (size_t)(16 + lr) * T_SEQ + kt + hi * 16);
        o1 = wmma_bf16(pa, b, o1);
        b  = load_frag_b(Vh + (size_t)(32 + lr) * T_SEQ + kt + hi * 16);
        o2 = wmma_bf16(pa, b, o2);
        b  = load_frag_b(Vh + (size_t)(48 + lr) * T_SEQ + kt + hi * 16);
        o3 = wmma_bf16(pa, b, o3);
    }

    // Normalize and store context in [B,T,D] bf16 for the output projection.
    const int b_ = bh / HEADS, h = bh % HEADS;
#pragma unroll
    for (int r = 0; r < 8; ++r) {
        float inv = 1.0f / lrow[r];
        int trow  = q0 + r + hi * 8;
        size_t base = ((size_t)(b_ * T_SEQ + trow)) * D_IN + h * DH;
        Ctx[base + 0  + lr] = (bf16_t)(o0[r] * inv);
        Ctx[base + 16 + lr] = (bf16_t)(o1[r] * inv);
        Ctx[base + 32 + lr] = (bf16_t)(o2[r] * inv);
        Ctx[base + 48 + lr] = (bf16_t)(o3[r] * inv);
    }
}

// ----------------------------- launcher ------------------------------------

extern "C" void kernel_launch(void* const* d_in, const int* in_sizes, int n_in,
                              void* d_out, int out_size, void* d_ws, size_t ws_size,
                              hipStream_t stream) {
    (void)in_sizes; (void)n_in; (void)out_size; (void)ws_size;

    const float* x  = (const float*)d_in[0];
    const float* wq = (const float*)d_in[1];
    const float* wk = (const float*)d_in[2];
    const float* wv = (const float*)d_in[3];
    const float* wo = (const float*)d_in[4];
    float* out = (float*)d_out;

    const size_t nX = (size_t)MROWS * D_IN;      // 5,242,880 elems
    const size_t nW = (size_t)D_IN * D_IN;       // 1,638,400 elems
    // Workspace layout (bf16 each): total ~62.5 MB.
    char* ws = (char*)d_ws;
    size_t off = 0;
    bf16_t* Xb  = (bf16_t*)(ws + off); off += nX * 2;
    bf16_t* WqT = (bf16_t*)(ws + off); off += nW * 2;
    bf16_t* WkT = (bf16_t*)(ws + off); off += nW * 2;
    bf16_t* WvT = (bf16_t*)(ws + off); off += nW * 2;
    bf16_t* WoT = (bf16_t*)(ws + off); off += nW * 2;
    bf16_t* Qb  = (bf16_t*)(ws + off); off += nX * 2;   // [B,H,T,dh]
    bf16_t* Kb  = (bf16_t*)(ws + off); off += nX * 2;   // [B,H,T,dh]
    bf16_t* Vt  = (bf16_t*)(ws + off); off += nX * 2;   // [B,H,dh,T]
    bf16_t* Ctx = (bf16_t*)(ws + off); off += nX * 2;   // [B,T,D]

    // 1) precision conversion / weight transposition
    cvt_f32_to_bf16<<<(int)(nX / 4 / 256), 256, 0, stream>>>(x, Xb, (int)nX);
    transpose_to_bf16<<<(int)(nW / 256), 256, 0, stream>>>(wq, WqT, D_IN, D_IN);
    transpose_to_bf16<<<(int)(nW / 256), 256, 0, stream>>>(wk, WkT, D_IN, D_IN);
    transpose_to_bf16<<<(int)(nW / 256), 256, 0, stream>>>(wv, WvT, D_IN, D_IN);
    transpose_to_bf16<<<(int)(nW / 256), 256, 0, stream>>>(wo, WoT, D_IN, D_IN);

    // 2) QKV projections (WMMA GEMM), layouts chosen for the attention kernel
    dim3 ggrid(D_IN / 128, MROWS / 128);   // (10, 32)
    gemm_bf16_wmma<<<ggrid, 256, 0, stream>>>(Xb, WqT, Qb, MROWS, D_IN, D_IN, 0);
    gemm_bf16_wmma<<<ggrid, 256, 0, stream>>>(Xb, WkT, Kb, MROWS, D_IN, D_IN, 0);
    gemm_bf16_wmma<<<ggrid, 256, 0, stream>>>(Xb, WvT, Vt, MROWS, D_IN, D_IN, 1);

    // 3) flash attention
    attn_kernel<<<BATCH * HEADS * (T_SEQ / 128), 256, 0, stream>>>(Qb, Kb, Vt, Ctx);

    // 4) output projection -> f32 d_out
    gemm_bf16_wmma<<<ggrid, 256, 0, stream>>>(Ctx, WoT, out, MROWS, D_IN, D_IN, 2);
}
